// ClassModel_72318659330833
// MI455X (gfx1250) — matrix operand order repaired and checked by reading, not compiled, not run on hardware
//
#include <hip/hip_runtime.h>
#include <hip/hip_bf16.h>
#include <math.h>

#define B_  256
#define S_  128
#define D_  768
#define K2_ 1536   // 2*D
#define MASK_ID_ 103

typedef __attribute__((ext_vector_type(16))) _Float16 v16h;
typedef __attribute__((ext_vector_type(8)))  _Float16 v8h;
typedef __attribute__((ext_vector_type(8)))  float    v8f;

// 21 label ids: groups of 8, 6, 7
__constant__ int c_lbl[21] = {
    2307, 2204, 3835, 2157, 6581, 2986, 5151, 3893,
    7929, 24791, 8699, 4257, 16021, 6623,
    6659, 2919, 11771, 3532, 11325, 4997, 13135};

// ---------------------------------------------------------------------------
// Kernel 1: f32 -> f16 weight conversion for the WMMA dense layer
// ---------------------------------------------------------------------------
__global__ void cvt_f16_kernel(const float* __restrict__ src,
                               _Float16* __restrict__ dst, int n) {
    int i = blockIdx.x * 256 + threadIdx.x;
    if (i < n) dst[i] = (_Float16)src[i];
}

// ---------------------------------------------------------------------------
// Kernel 2: per-batch-row attention pooling + category head + feats (f16)
// One block (256 threads = 8 waves) per batch row b.
// ---------------------------------------------------------------------------
__global__ void attn_feats_kernel(const float* __restrict__ bert_out,
                                  const int*   __restrict__ input_ids,
                                  const int*   __restrict__ length,
                                  const float* __restrict__ senti_w,
                                  const float* __restrict__ senti_b,
                                  _Float16*    __restrict__ feats16,  // (B, 2D)
                                  float*       __restrict__ cat_out)  // (B, 2)
{
    __shared__ float ml[D_];     // mask_logits row
    __shared__ float sc[S_];     // scores
    __shared__ float red[256];
    __shared__ int   s_pos;

    const int b    = blockIdx.x;
    const int tid  = threadIdx.x;
    const int lane = tid & 31;
    const int wave = tid >> 5;

    if (tid == 0) s_pos = S_;
    __syncthreads();
    if (tid < S_) {
        if (input_ids[b * S_ + tid] == MASK_ID_) atomicMin(&s_pos, tid);
    }
    __syncthreads();
    const int mp = (s_pos == S_) ? 0 : s_pos;   // argmax semantics: 0 if none

    // mask_logits = bert_out[b, mp, :]
    const float* rowm = bert_out + ((size_t)b * S_ + mp) * D_;
    for (int d = tid; d < D_; d += 256) ml[d] = rowm[d];
    __syncthreads();

    // scores_raw[s] = dot(bert_out[b,s,:], ml)   (wave per s, coalesced in d)
    for (int s = wave; s < S_; s += 8) {
        const float* rs = bert_out + ((size_t)b * S_ + s) * D_;
        float acc = 0.f;
        for (int d = lane; d < D_; d += 32) acc += rs[d] * ml[d];
        #pragma unroll
        for (int off = 16; off > 0; off >>= 1) acc += __shfl_xor(acc, off, 32);
        if (lane == 0) sc[s] = acc;
    }
    __syncthreads();

    // masked softmax over s in [3, 3+len)
    const int len = length[b];
    const bool valid = (tid < S_) && (tid >= 3) && (tid < 3 + len);
    float v = valid ? sc[tid] : -INFINITY;
    red[tid] = v;
    __syncthreads();
    for (int o = 128; o > 0; o >>= 1) {
        if (tid < o) red[tid] = fmaxf(red[tid], red[tid + o]);
        __syncthreads();
    }
    const float mx = red[0];
    __syncthreads();
    float e = valid ? __expf(sc[tid] - mx) : 0.f;
    red[tid] = e;
    __syncthreads();
    for (int o = 128; o > 0; o >>= 1) {
        if (tid < o) red[tid] += red[tid + o];
        __syncthreads();
    }
    const float ssum = red[0];
    __syncthreads();
    if (tid < S_) sc[tid] = e / ssum;
    __syncthreads();

    // att[d] = sum_s sc[s] * bert_out[b,s,d]; feats = [att | mask_logits] (f16)
    for (int d = tid; d < D_; d += 256) {
        float a = 0.f;
        const float* base = bert_out + (size_t)b * S_ * D_ + d;
        for (int s = 0; s < S_; ++s) a += sc[s] * base[(size_t)s * D_];
        feats16[(size_t)b * K2_ + d]       = (_Float16)a;
        feats16[(size_t)b * K2_ + D_ + d]  = (_Float16)ml[d];
    }

    // category_out = pooler @ senti_w.T + senti_b   (pooler = bert_out[b,0,:])
    const float* pool = bert_out + (size_t)b * S_ * D_;
    float p0 = 0.f, p1 = 0.f;
    for (int d = tid; d < D_; d += 256) {
        const float pv = pool[d];
        p0 += pv * senti_w[d];
        p1 += pv * senti_w[D_ + d];
    }
    __syncthreads();
    red[tid] = p0;
    __syncthreads();
    for (int o = 128; o > 0; o >>= 1) {
        if (tid < o) red[tid] += red[tid + o];
        __syncthreads();
    }
    const float c0 = red[0];
    __syncthreads();
    red[tid] = p1;
    __syncthreads();
    for (int o = 128; o > 0; o >>= 1) {
        if (tid < o) red[tid] += red[tid + o];
        __syncthreads();
    }
    const float c1 = red[0];
    if (tid == 0) {
        cat_out[b * 2 + 0] = c0 + senti_b[0];
        cat_out[b * 2 + 1] = c1 + senti_b[1];
    }
}

// ---------------------------------------------------------------------------
// Kernel 3: h = tanh(feats @ dense_w.T + dense_b) via v_wmma_f32_16x16x32_f16
// M=256, N=768, K=1536. One wave per 16x16 output tile; 768 tiles total.
// Grid: 96 blocks x 256 threads (8 waves/block). Fully convergent -> EXEC all 1s.
// ---------------------------------------------------------------------------
__global__ void dense_wmma_kernel(const _Float16* __restrict__ A,   // (256,1536) feats f16
                                  const _Float16* __restrict__ Bm,  // (768,1536) dense_w f16
                                  const float*    __restrict__ bias,// (768,)
                                  float*          __restrict__ H)   // (256,768)
{
    const int tid  = threadIdx.x;
    const int lane = tid & 31;
    const int wave = tid >> 5;
    const int tile = blockIdx.x * 8 + wave;     // 0..767
    const int mt   = tile / 48;                 // 16 M-tiles
    const int nt   = tile % 48;                 // 48 N-tiles
    const int hi   = (lane >= 16) ? 1 : 0;
    const int l15  = lane & 15;

    const _Float16* arow = A  + (size_t)(mt * 16 + l15) * K2_;
    const _Float16* brow = Bm + (size_t)(nt * 16 + l15) * K2_;

    v8f acc = {};
    for (int k0 = 0; k0 < K2_; k0 += 32) {
        // A fragment (16x32 f16): lanes<16 rows M=lane, K k0..k0+15;
        // lanes>=16 rows M=lane-16, K k0+16..k0+31
        const int ka = k0 + (hi ? 16 : 0);
        const v8h alo = *(const v8h*)(arow + ka);
        const v8h ahi = *(const v8h*)(arow + ka + 8);
        v16h a;
        #pragma unroll
        for (int i = 0; i < 8; ++i) { a[i] = alo[i]; a[i + 8] = ahi[i]; }

        // B fragment (32x16 f16): lane holds column N=lane%16;
        // lanes<16: K=k0..k0+15, lanes>=16: K=k0+16..k0+31 (contiguous in dense_w row)
        const int kb = k0 + (hi ? 16 : 0);
        const v16h bf = *(const v16h*)(brow + kb);

        acc = __builtin_amdgcn_wmma_f32_16x16x32_f16(
            /*neg_a=*/false, a, /*neg_b=*/false, bf,
            /*c_mod=*/(short)0, acc, /*reuse_a=*/false, /*reuse_b=*/false);
    }

    // Epilogue: C/D layout — VGPR r: lanes<16 -> M=r, N=lane; lanes>=16 -> M=r+8, N=lane-16
    const int n = nt * 16 + l15;
    const float bn = bias[n];
    #pragma unroll
    for (int r = 0; r < 8; ++r) {
        const int m = mt * 16 + r + (hi ? 8 : 0);
        H[(size_t)m * D_ + n] = tanhf(acc[r] + bn);
    }
}

// ---------------------------------------------------------------------------
// Kernel 4: only the 21 needed vocab columns of the decoder + label projections
// One block (8 waves) per batch row.
// ---------------------------------------------------------------------------
__global__ void head_kernel(const float* __restrict__ H,     // (256,768)
                            const float* __restrict__ dec_w, // (V,768)
                            const float* __restrict__ dec_b, // (V,)
                            const float* __restrict__ w0,    // (2,8)
                            const float* __restrict__ w1,    // (2,6)
                            const float* __restrict__ w2,    // (2,7)
                            float*       __restrict__ out)   // (B,2,3)
{
    __shared__ float p[21];
    const int b    = blockIdx.x;
    const int tid  = threadIdx.x;
    const int lane = tid & 31;
    const int wave = tid >> 5;
    const float* hrow = H + (size_t)b * D_;

    for (int j = wave; j < 21; j += 8) {
        const int id = c_lbl[j];
        const float* dr = dec_w + (size_t)id * D_;
        float acc = 0.f;
        for (int d = lane; d < D_; d += 32) acc += hrow[d] * dr[d];
        #pragma unroll
        for (int off = 16; off > 0; off >>= 1) acc += __shfl_xor(acc, off, 32);
        if (lane == 0) p[j] = tanhf(acc + dec_b[id]);
    }
    __syncthreads();

    if (tid == 0) {
        const float* wg[3] = {w0, w1, w2};
        const int    go[4] = {0, 8, 14, 21};
        for (int g = 0; g < 3; ++g) {
            const int cnt = go[g + 1] - go[g];
            for (int c = 0; c < 2; ++c) {
                float acc = 0.f;
                for (int j = 0; j < cnt; ++j) acc += p[go[g] + j] * wg[g][c * cnt + j];
                out[b * 6 + c * 3 + g] = acc;   // (B,2,3) row-major
            }
        }
    }
}

// ---------------------------------------------------------------------------
extern "C" void kernel_launch(void* const* d_in, const int* in_sizes, int n_in,
                              void* d_out, int out_size, void* d_ws, size_t ws_size,
                              hipStream_t stream) {
    const float* bert_out  = (const float*)d_in[0];
    const int*   input_ids = (const int*)  d_in[1];
    const int*   length    = (const int*)  d_in[2];
    const float* senti_w   = (const float*)d_in[3];
    const float* senti_b   = (const float*)d_in[4];
    const float* dense_w   = (const float*)d_in[5];
    const float* dense_b   = (const float*)d_in[6];
    const float* dec_w     = (const float*)d_in[7];
    const float* dec_b     = (const float*)d_in[8];
    const float* w0        = (const float*)d_in[9];
    const float* w1        = (const float*)d_in[10];
    const float* w2        = (const float*)d_in[11];

    float* cat_out = (float*)d_out;          // 256*2
    float* out2    = (float*)d_out + B_ * 2; // 256*2*3

    // workspace layout
    char* ws = (char*)d_ws;
    _Float16* dw16    = (_Float16*)ws;                                   // 768*1536 f16
    _Float16* feats16 = (_Float16*)(ws + (size_t)D_ * K2_ * 2);          // 256*1536 f16
    float*    H       = (float*)   (ws + (size_t)D_ * K2_ * 2
                                       + (size_t)B_ * K2_ * 2);          // 256*768 f32

    const int ndw = D_ * K2_;
    cvt_f16_kernel<<<(ndw + 255) / 256, 256, 0, stream>>>(dense_w, dw16, ndw);

    attn_feats_kernel<<<B_, 256, 0, stream>>>(bert_out, input_ids, length,
                                              senti_w, senti_b, feats16, cat_out);

    dense_wmma_kernel<<<96, 256, 0, stream>>>(feats16, dw16, dense_b, H);

    head_kernel<<<B_, 256, 0, stream>>>(H, dec_w, dec_b, w0, w1, w2, out2);
}
